// MultiTaskLongformerModel_14413910245711
// MI455X (gfx1250) — matrix operand order repaired and checked by reading, not compiled, not run on hardware
//
#include <hip/hip_runtime.h>

#define S_  4096
#define D_  768
#define H_  12
#define DH_ 64
#define L_  12
#define FF_ 3072
#define WONE 256

// Flip to 0 if the async-to-LDS mnemonic fails to assemble.
#define GEMM_ASYNC 1

typedef __bf16 v16bf __attribute__((ext_vector_type(16)));
typedef float  v8f   __attribute__((ext_vector_type(8)));

struct alignas(16) U4 { unsigned x, y, z, w; };
union AF { v16bf v; U4 q[2]; };

__device__ __forceinline__ v8f zero8() {
  v8f z = {0.f,0.f,0.f,0.f,0.f,0.f,0.f,0.f};
  return z;
}

__device__ __forceinline__ unsigned short f2bf(float f) {
  union { float f; unsigned u; } x; x.f = f;
  unsigned r = x.u + 0x7FFFu + ((x.u >> 16) & 1u);   // RNE f32 -> bf16
  return (unsigned short)(r >> 16);
}

__device__ __forceinline__ v8f wmma_bf16(const AF& a, const AF& b, v8f c) {
  return __builtin_amdgcn_wmma_f32_16x16x32_bf16(false, a.v, false, b.v,
                                                 (short)0, c, false, false);
}

#if GEMM_ASYNC
// Per-lane async copy of 16B global -> LDS. Generic pointers to __shared__ are
// (aperture_hi | lds_offset), so the low 32 bits are the wave-relative LDS
// byte offset the instruction's VDST operand expects.
__device__ __forceinline__ void async_ld16(unsigned ldsOff, const unsigned short* g) {
  asm volatile("global_load_async_to_lds_b128 %0, %1, off"
               :: "v"(ldsOff), "v"((unsigned long long)(size_t)g)
               : "memory");
}
__device__ __forceinline__ void wait_async0() {
  asm volatile("s_wait_asynccnt 0" ::: "memory");
}
#endif

// ---------------------------------------------------------------- pos ids
__global__ __launch_bounds__(256) void posid_kernel(const int* __restrict__ mask,
                                                    int* __restrict__ pos) {
  __shared__ int part[256];
  int t = threadIdx.x;
  int base = t * 16;
  int loc[16];
  int s = 0;
  for (int i = 0; i < 16; i++) { s += mask[base + i]; loc[i] = s; }
  part[t] = s;
  __syncthreads();
  for (int off = 1; off < 256; off <<= 1) {
    int v = (t >= off) ? part[t - off] : 0;
    __syncthreads();
    part[t] += v;
    __syncthreads();
  }
  int prev = (t == 0) ? 0 : part[t - 1];
  for (int i = 0; i < 16; i++) {
    int c = prev + loc[i];
    pos[base + i] = c * mask[base + i] + 1;
  }
}

// ---------------------------------------------------------------- reductions
__device__ __forceinline__ void blk_reduce2(float& a, float& b) {
  __shared__ float ra[256], rb[256];
  int t = threadIdx.x;
  ra[t] = a; rb[t] = b;
  __syncthreads();
  for (int off = 128; off > 0; off >>= 1) {
    if (t < off) { ra[t] += ra[t + off]; rb[t] += rb[t + off]; }
    __syncthreads();
  }
  a = ra[0]; b = rb[0];
  __syncthreads();
}

// ---------------------------------------------------------------- embeddings + LN
__global__ __launch_bounds__(256) void embed_ln_kernel(
    const int* __restrict__ ids, const int* __restrict__ pos,
    const float* __restrict__ ew, const float* __restrict__ ep,
    const float* __restrict__ et, const float* __restrict__ g,
    const float* __restrict__ b, float* __restrict__ x,
    unsigned short* __restrict__ xb) {
  int s = blockIdx.x, t = threadIdx.x;
  const float* w = ew + (size_t)ids[s] * D_;
  const float* p = ep + (size_t)pos[s] * D_;
  float v[3], sum = 0.f, sq = 0.f;
  for (int i = 0; i < 3; i++) {
    int d = t + i * 256;
    float val = w[d] + p[d] + et[d];
    v[i] = val; sum += val; sq += val * val;
  }
  blk_reduce2(sum, sq);
  float mu = sum * (1.f / D_);
  float var = sq * (1.f / D_) - mu * mu;
  float rs = rsqrtf(var + 1e-5f);
  for (int i = 0; i < 3; i++) {
    int d = t + i * 256;
    float o = (v[i] - mu) * rs * g[d] + b[d];
    x[(size_t)s * D_ + d] = o;
    xb[(size_t)s * D_ + d] = f2bf(o);
  }
}

// ---------------------------------------------------------------- residual + LN
__global__ __launch_bounds__(256) void ln_residual_kernel(
    float* __restrict__ x, const float* __restrict__ y,
    const float* __restrict__ g, const float* __restrict__ b,
    unsigned short* __restrict__ xb) {
  int s = blockIdx.x, t = threadIdx.x;
  float v[3], sum = 0.f, sq = 0.f;
  for (int i = 0; i < 3; i++) {
    int d = t + i * 256;
    float val = x[(size_t)s * D_ + d] + y[(size_t)s * D_ + d];
    v[i] = val; sum += val; sq += val * val;
  }
  blk_reduce2(sum, sq);
  float mu = sum * (1.f / D_);
  float var = sq * (1.f / D_) - mu * mu;
  float rs = rsqrtf(var + 1e-5f);
  for (int i = 0; i < 3; i++) {
    int d = t + i * 256;
    float o = (v[i] - mu) * rs * g[d] + b[d];
    x[(size_t)s * D_ + d] = o;
    xb[(size_t)s * D_ + d] = f2bf(o);
  }
}

// ---------------------------------------------------------------- transpose f32 -> bf16
// dst[n*K + k] = bf16(src[k*N + n]);  grid = (K/32, N/32), block = 256
__global__ __launch_bounds__(256) void transpose_bf16_kernel(
    const float* __restrict__ src, unsigned short* __restrict__ dst,
    int K, int N) {
  __shared__ unsigned short tl[32][40];
  int k0 = blockIdx.x * 32, n0 = blockIdx.y * 32;
  int t = threadIdx.x;
  int c = t & 31, r = t >> 5;   // r in [0,8)
  for (int rr = 0; rr < 4; rr++) {
    int k = r + rr * 8;
    tl[c][k] = f2bf(src[(size_t)(k0 + k) * N + n0 + c]);
  }
  __syncthreads();
  for (int rr = 0; rr < 4; rr++) {
    int n = r + rr * 8;
    dst[(size_t)(n0 + n) * K + k0 + c] = tl[n][c];
  }
}

// ---------------------------------------------------------------- WMMA GEMM
// C(M,N) = A(M,K) * Bt(N,K)^T + bias;  block 256 (8 waves), tile 128x128,
// wave computes 32x64 (8 independent WMMA chains). Double-buffered LDS:
// the async tile fill for step k+1 overlaps the WMMAs of step k.
#define MODE_F32   0
#define MODE_BF16  1
#define MODE_GELU  2
#define MODE_HEADS 3
#define MODE_VT    4

#define TILE_E 5120u              // elements per buffer (128 rows * 40)
#define TILE_B 10240u             // bytes per buffer

__global__ __launch_bounds__(256) void gemm_bf16_kernel(
    const unsigned short* __restrict__ A, const unsigned short* __restrict__ Bt,
    const float* __restrict__ bias, float* __restrict__ outf,
    unsigned short* __restrict__ outb, int M, int N, int K, int mode) {
  __shared__ unsigned short sA[2 * TILE_E];
  __shared__ unsigned short sB[2 * TILE_E];
  int tid = threadIdx.x;
  int m0 = blockIdx.x * 128, n0 = blockIdx.y * 128;
  int w = tid >> 5, lane = tid & 31, l = lane & 15, hi = lane >> 4;
  int wm = w & 3, wn = w >> 2;       // wave tile: rows [wm*32,+32), cols [wn*64,+64)

  v8f acc[2][4];
  for (int i = 0; i < 2; i++) for (int j = 0; j < 4; j++) acc[i][j] = zero8();

  int row0 = tid >> 2, cc = tid & 3;   // chunk c=tid    -> rows 0..63
  int row1 = row0 + 64;                // chunk c=tid+256 -> rows 64..127

  const unsigned short* gA0 = A + (size_t)(m0 + row0) * K + cc * 8;
  const unsigned short* gA1 = A + (size_t)(m0 + row1) * K + cc * 8;
  const unsigned short* gB0 = Bt + (size_t)(n0 + row0) * K + cc * 8;
  const unsigned short* gB1 = Bt + (size_t)(n0 + row1) * K + cc * 8;
#if GEMM_ASYNC
  unsigned oA0 = (unsigned)(size_t)&sA[row0 * 40 + cc * 8];
  unsigned oA1 = (unsigned)(size_t)&sA[row1 * 40 + cc * 8];
  unsigned oB0 = (unsigned)(size_t)&sB[row0 * 40 + cc * 8];
  unsigned oB1 = (unsigned)(size_t)&sB[row1 * 40 + cc * 8];
  // prologue: fill buffer 0 with the k=0 tile
  async_ld16(oA0, gA0);
  async_ld16(oA1, gA1);
  async_ld16(oB0, gB0);
  async_ld16(oB1, gB1);
#endif

  int cur = 0;
  for (int k0 = 0; k0 < K; k0 += 32) {
#if GEMM_ASYNC
    wait_async0();                 // tile for this step has landed in LDS
    __syncthreads();
    if (k0 + 32 < K) {             // kick off next tile into the other buffer
      unsigned boff = cur ? 0u : TILE_B;
      async_ld16(oA0 + boff, gA0 + k0 + 32);
      async_ld16(oA1 + boff, gA1 + k0 + 32);
      async_ld16(oB0 + boff, gB0 + k0 + 32);
      async_ld16(oB1 + boff, gB1 + k0 + 32);
      __builtin_prefetch((const void*)(gA0 + k0 + 64), 0, 1);
      __builtin_prefetch((const void*)(gB0 + k0 + 64), 0, 1);
    }
#else
    __syncthreads();
    *(U4*)&sA[cur * TILE_E + row0 * 40 + cc * 8] = *(const U4*)(gA0 + k0);
    *(U4*)&sA[cur * TILE_E + row1 * 40 + cc * 8] = *(const U4*)(gA1 + k0);
    *(U4*)&sB[cur * TILE_E + row0 * 40 + cc * 8] = *(const U4*)(gB0 + k0);
    *(U4*)&sB[cur * TILE_E + row1 * 40 + cc * 8] = *(const U4*)(gB1 + k0);
    __syncthreads();
#endif
    const unsigned short* bA = &sA[cur * TILE_E];
    const unsigned short* bB = &sB[cur * TILE_E];

    AF a[2], b[4];
    for (int i = 0; i < 2; i++) {
      const U4* p = (const U4*)&bA[(wm * 32 + i * 16 + l) * 40 + hi * 8];
      a[i].q[0] = p[0]; a[i].q[1] = p[2];
    }
    for (int j = 0; j < 4; j++) {
      const U4* p = (const U4*)&bB[(wn * 64 + j * 16 + l) * 40 + hi * 8];
      b[j].q[0] = p[0]; b[j].q[1] = p[2];
    }
    for (int i = 0; i < 2; i++)
      for (int j = 0; j < 4; j++)
        acc[i][j] = wmma_bf16(a[i], b[j], acc[i][j]);
    cur ^= 1;
  }

  for (int i = 0; i < 2; i++)
    for (int j = 0; j < 4; j++)
      for (int e = 0; e < 8; e++) {
        int row = m0 + wm * 32 + i * 16 + e + hi * 8;
        int col = n0 + wn * 64 + j * 16 + l;
        float val = acc[i][j][e] + bias[col];
        if (mode == MODE_F32) {
          outf[(size_t)row * N + col] = val;
        } else if (mode == MODE_BF16) {
          outb[(size_t)row * N + col] = f2bf(val);
        } else if (mode == MODE_GELU) {
          val = 0.5f * val * (1.0f + erff(val * 0.70710678118f));
          outb[(size_t)row * N + col] = f2bf(val);
        } else if (mode == MODE_HEADS) {       // [h][s][dh]
          int h = col >> 6, dh = col & 63;
          outb[((size_t)h * M + row) * 64 + dh] = f2bf(val);
        } else {                                // MODE_VT: [h][dh][s] == col-major
          outb[(size_t)col * M + row] = f2bf(val);
        }
      }
}

// ---------------------------------------------------------------- sliding attention
// grid (nb=16, H=12), block 256 (8 waves); wave handles 32 query rows.
__global__ __launch_bounds__(256) void attn_kernel(
    const unsigned short* __restrict__ qb, const unsigned short* __restrict__ kb,
    const unsigned short* __restrict__ vb, const int* __restrict__ amask,
    unsigned short* __restrict__ aout) {
  __shared__ unsigned short sP[8 * 16 * 40];
  int blk = blockIdx.x, h = blockIdx.y;
  int tid = threadIdx.x, w = tid >> 5, lane = tid & 31;
  int l = lane & 15, hi = lane >> 4;
  int r0 = blk * 256 + w * 32;
  const unsigned short* qh = qb + (size_t)h * S_ * 64;
  const unsigned short* kh = kb + (size_t)h * S_ * 64;
  const unsigned short* vh = vb + (size_t)h * 64 * S_;
  unsigned short* myP = &sP[w * 16 * 40];

  AF aq[2][2];
  for (int i = 0; i < 2; i++) {
    int row = r0 + i * 16 + l;
    for (int t = 0; t < 2; t++) {
      const U4* p = (const U4*)(qh + (size_t)row * 64 + t * 32 + hi * 8);
      aq[i][t].q[0] = p[0]; aq[i][t].q[1] = p[2];
    }
  }

  v8f oacc[2][4];
  for (int i = 0; i < 2; i++) for (int n = 0; n < 4; n++) oacc[i][n] = zero8();
  float m[2][8], lsum[2][8];
  for (int i = 0; i < 2; i++) for (int e = 0; e < 8; e++) { m[i][e] = -1e30f; lsum[i][e] = 0.f; }

  int kbase = blk * 256 - 256;
  for (int kt = 0; kt < 24; kt++) {
    int ks = kbase + kt * 32;
    if (ks < 0 || ks + 32 > S_) continue;                 // padding tiles (32-aligned)
    if (ks > r0 + 31 + WONE || ks + 31 < r0 - WONE) continue;  // outside band

    AF bk[2][2];
    for (int j = 0; j < 2; j++) {
      int key = ks + j * 16 + l;
      for (int t = 0; t < 2; t++) {
        const U4* p = (const U4*)(kh + (size_t)key * 64 + t * 32 + hi * 8);
        bk[j][t].q[0] = p[0]; bk[j][t].q[1] = p[2];
      }
    }
    int key0 = ks + l, key1 = ks + 16 + l;
    int mk0 = amask[key0], mk1 = amask[key1];

    for (int i = 0; i < 2; i++) {
      v8f s0 = zero8(), s1 = zero8();
      s0 = wmma_bf16(aq[i][0], bk[0][0], s0);
      s0 = wmma_bf16(aq[i][1], bk[0][1], s0);
      s1 = wmma_bf16(aq[i][0], bk[1][0], s1);
      s1 = wmma_bf16(aq[i][1], bk[1][1], s1);

      float p0[8], p1[8];
      for (int e = 0; e < 8; e++) {
        int row = r0 + i * 16 + e + hi * 8;
        int rel0 = key0 - row, rel1 = key1 - row;
        float v0 = (mk0 && rel0 >= -WONE && rel0 <= WONE) ? s0[e] * 0.125f : -1e9f;
        float v1 = (mk1 && rel1 >= -WONE && rel1 <= WONE) ? s1[e] * 0.125f : -1e9f;
        float tmax = fmaxf(v0, v1);
        for (int d = 1; d < 16; d <<= 1) tmax = fmaxf(tmax, __shfl_xor(tmax, d, 32));
        float mn = fmaxf(m[i][e], tmax);
        float corr = __expf(m[i][e] - mn);
        m[i][e] = mn;
        float e0 = __expf(v0 - mn), e1 = __expf(v1 - mn);
        float rs = e0 + e1;
        for (int d = 1; d < 16; d <<= 1) rs += __shfl_xor(rs, d, 32);
        lsum[i][e] = lsum[i][e] * corr + rs;
        for (int n = 0; n < 4; n++) oacc[i][n][e] *= corr;
        p0[e] = e0; p1[e] = e1;
      }
      // re-fragment probabilities via wave-private LDS
      for (int e = 0; e < 8; e++) {
        int rl = e + hi * 8;
        myP[rl * 40 + l] = f2bf(p0[e]);
        myP[rl * 40 + 16 + l] = f2bf(p1[e]);
      }
      AF ap;
      const U4* pp = (const U4*)&myP[l * 40 + hi * 8];
      ap.q[0] = pp[0]; ap.q[1] = pp[2];
      for (int n = 0; n < 4; n++) {
        int dh = n * 16 + l;
        const U4* pv = (const U4*)(vh + (size_t)dh * S_ + ks + hi * 8);
        AF bv; bv.q[0] = pv[0]; bv.q[1] = pv[2];
        oacc[i][n] = wmma_bf16(ap, bv, oacc[i][n]);
      }
    }
  }

  for (int i = 0; i < 2; i++)
    for (int n = 0; n < 4; n++)
      for (int e = 0; e < 8; e++) {
        int row = r0 + i * 16 + e + hi * 8;
        int dh = n * 16 + l;
        float val = oacc[i][n][e] / lsum[i][e];
        aout[(size_t)row * D_ + h * 64 + dh] = f2bf(val);
      }
}

// ---------------------------------------------------------------- classifier heads
__global__ __launch_bounds__(256) void cls_hidden_kernel(
    const float* __restrict__ x, const float* __restrict__ cW1,
    const float* __restrict__ cb1, const float* __restrict__ dW1,
    const float* __restrict__ db1, float* __restrict__ hid) {
  const float* W = blockIdx.x ? dW1 : cW1;
  const float* bb = blockIdx.x ? db1 : cb1;
  float* out = hid + blockIdx.x * 512;
  int t = threadIdx.x;
  for (int rep = 0; rep < 2; rep++) {
    int j = t + rep * 256;
    float s = bb[j];
    for (int k = 0; k < D_; k++) s += x[k] * W[k * 512 + j];
    out[j] = fmaxf(s, 0.f);
  }
}

__global__ void cls_out_kernel(const float* __restrict__ hid,
                               const float* __restrict__ cW2, const float* __restrict__ cb2,
                               const float* __restrict__ dW2, const float* __restrict__ db2,
                               float* __restrict__ out) {
  int j = threadIdx.x;
  if (j < 5) {
    float s = cb2[j];
    for (int k = 0; k < 512; k++) s += hid[k] * cW2[k * 5 + j];
    out[j] = s;
  } else if (j < 15) {
    int jj = j - 5;
    float s = db2[jj];
    for (int k = 0; k < 512; k++) s += hid[512 + k] * dW2[k * 10 + jj];
    out[j] = s;
  }
}

// ---------------------------------------------------------------- launch
extern "C" void kernel_launch(void* const* d_in, const int* in_sizes, int n_in,
                              void* d_out, int out_size, void* d_ws, size_t ws_size,
                              hipStream_t stream) {
  (void)in_sizes; (void)n_in; (void)out_size; (void)ws_size;
  const int*   ids    = (const int*)d_in[0];
  const int*   amask  = (const int*)d_in[1];
  const float* ew     = (const float*)d_in[2];
  const float* ep     = (const float*)d_in[3];
  const float* et     = (const float*)d_in[4];
  const float* elg    = (const float*)d_in[5];
  const float* elb    = (const float*)d_in[6];
  const float* Wq     = (const float*)d_in[7];
  const float* bq     = (const float*)d_in[8];
  const float* Wk     = (const float*)d_in[9];
  const float* bk     = (const float*)d_in[10];
  const float* Wv     = (const float*)d_in[11];
  const float* bv     = (const float*)d_in[12];
  const float* Wo     = (const float*)d_in[13];
  const float* bo     = (const float*)d_in[14];
  const float* ln1g   = (const float*)d_in[15];
  const float* ln1b   = (const float*)d_in[16];
  const float* W1     = (const float*)d_in[17];
  const float* b1     = (const float*)d_in[18];
  const float* W2     = (const float*)d_in[19];
  const float* b2     = (const float*)d_in[20];
  const float* ln2g   = (const float*)d_in[21];
  const float* ln2b   = (const float*)d_in[22];
  const float* cW1    = (const float*)d_in[23];
  const float* cb1    = (const float*)d_in[24];
  const float* cW2    = (const float*)d_in[25];
  const float* cb2    = (const float*)d_in[26];
  const float* dW1    = (const float*)d_in[27];
  const float* db1    = (const float*)d_in[28];
  const float* dW2    = (const float*)d_in[29];
  const float* db2    = (const float*)d_in[30];
  float* outp = (float*)d_out;

  char* wsb = (char*)d_ws;
  size_t off = 0;
  auto alloc = [&](size_t bytes) -> char* {
    char* r = wsb + off;
    off = (off + bytes + 255) & ~(size_t)255;
    return r;
  };
  int*            pos  = (int*)alloc((size_t)S_ * 4);
  float*          x    = (float*)alloc((size_t)S_ * D_ * 4);
  unsigned short* xb   = (unsigned short*)alloc((size_t)S_ * D_ * 2);
  unsigned short* qbuf = (unsigned short*)alloc((size_t)S_ * D_ * 2);
  unsigned short* kbuf = (unsigned short*)alloc((size_t)S_ * D_ * 2);
  unsigned short* vbuf = (unsigned short*)alloc((size_t)S_ * D_ * 2);
  unsigned short* abuf = (unsigned short*)alloc((size_t)S_ * D_ * 2);
  unsigned short* hb   = (unsigned short*)alloc((size_t)S_ * FF_ * 2);
  float*          gout = (float*)alloc((size_t)S_ * D_ * 4);
  unsigned short* wq_t = (unsigned short*)alloc((size_t)D_ * D_ * 2);
  unsigned short* wk_t = (unsigned short*)alloc((size_t)D_ * D_ * 2);
  unsigned short* wv_t = (unsigned short*)alloc((size_t)D_ * D_ * 2);
  unsigned short* wo_t = (unsigned short*)alloc((size_t)D_ * D_ * 2);
  unsigned short* w1_t = (unsigned short*)alloc((size_t)D_ * FF_ * 2);
  unsigned short* w2_t = (unsigned short*)alloc((size_t)FF_ * D_ * 2);
  float*          clsh = (float*)alloc(1024 * 4);

  posid_kernel<<<1, 256, 0, stream>>>(amask, pos);
  embed_ln_kernel<<<S_, 256, 0, stream>>>(ids, pos, ew, ep, et, elg, elb, x, xb);

  dim3 gD(S_ / 128, D_ / 128);
  dim3 gF(S_ / 128, FF_ / 128);
  dim3 tDD(D_ / 32, D_ / 32);
  dim3 tDF(D_ / 32, FF_ / 32);
  dim3 tFD(FF_ / 32, D_ / 32);

  for (int i = 0; i < L_; i++) {
    transpose_bf16_kernel<<<tDD, 256, 0, stream>>>(Wq + (size_t)i * D_ * D_, wq_t, D_, D_);
    transpose_bf16_kernel<<<tDD, 256, 0, stream>>>(Wk + (size_t)i * D_ * D_, wk_t, D_, D_);
    transpose_bf16_kernel<<<tDD, 256, 0, stream>>>(Wv + (size_t)i * D_ * D_, wv_t, D_, D_);
    transpose_bf16_kernel<<<tDD, 256, 0, stream>>>(Wo + (size_t)i * D_ * D_, wo_t, D_, D_);
    transpose_bf16_kernel<<<tDF, 256, 0, stream>>>(W1 + (size_t)i * D_ * FF_, w1_t, D_, FF_);
    transpose_bf16_kernel<<<tFD, 256, 0, stream>>>(W2 + (size_t)i * FF_ * D_, w2_t, FF_, D_);

    gemm_bf16_kernel<<<gD, 256, 0, stream>>>(xb, wq_t, bq + (size_t)i * D_, nullptr, qbuf, S_, D_, D_, MODE_HEADS);
    gemm_bf16_kernel<<<gD, 256, 0, stream>>>(xb, wk_t, bk + (size_t)i * D_, nullptr, kbuf, S_, D_, D_, MODE_HEADS);
    gemm_bf16_kernel<<<gD, 256, 0, stream>>>(xb, wv_t, bv + (size_t)i * D_, nullptr, vbuf, S_, D_, D_, MODE_VT);

    attn_kernel<<<dim3(S_ / 256, H_), 256, 0, stream>>>(qbuf, kbuf, vbuf, amask, abuf);

    gemm_bf16_kernel<<<gD, 256, 0, stream>>>(abuf, wo_t, bo + (size_t)i * D_, gout, nullptr, S_, D_, D_, MODE_F32);
    ln_residual_kernel<<<S_, 256, 0, stream>>>(x, gout, ln1g + (size_t)i * D_, ln1b + (size_t)i * D_, xb);

    gemm_bf16_kernel<<<gF, 256, 0, stream>>>(xb, w1_t, b1 + (size_t)i * FF_, nullptr, hb, S_, FF_, D_, MODE_GELU);
    gemm_bf16_kernel<<<gD, 256, 0, stream>>>(hb, w2_t, b2 + (size_t)i * D_, gout, nullptr, S_, D_, FF_, MODE_F32);
    ln_residual_kernel<<<S_, 256, 0, stream>>>(x, gout, ln2g + (size_t)i * D_, ln2b + (size_t)i * D_, xb);
  }

  cls_hidden_kernel<<<2, 256, 0, stream>>>(x, cW1, cb1, dW1, db1, clsh);
  cls_out_kernel<<<1, 32, 0, stream>>>(clsh, cW2, cb2, dW2, db2, outp);
}
